// SPMLP_41970420418169
// MI455X (gfx1250) — matrix operand order repaired and checked by reading, not compiled, not run on hardware
//
#include <hip/hip_runtime.h>
#include <hip/hip_bf16.h>
#include <cstdint>

// ---------------------------------------------------------------------------
// MoE MLP (top-2 of 8 experts), N=4096 tokens, D=1024, F=2816.
// bf16 WMMA (v_wmma_f32_16x16x32_bf16) with f32 accumulation; fp32 router.
// Sparse dispatch via per-expert compaction (142 GFLOP vs 580 dense).
// Weights stay fp32 in HBM (read once, ~277MB, ~12us at 23.3TB/s); converted
// to bf16 inline while staging to LDS. Per-expert weights (34.6MB) are
// L2-resident (192MB L2) across token tiles. Register-prefetch pipeline
// overlaps next tile's global loads with current tile's WMMAs.
// ---------------------------------------------------------------------------

#define N_TOK   4096
#define HID     1024
#define INTER   2816
#define NEXP    8
#define EF      (NEXP * INTER)   // 22528

typedef __attribute__((ext_vector_type(16))) __bf16 v16bf;
typedef __attribute__((ext_vector_type(8)))  __bf16 bf16x8;
typedef __attribute__((ext_vector_type(8)))  float  v8f;

__device__ __forceinline__ v16bf cat8(bf16x8 lo, bf16x8 hi) {
  v16bf r;
#pragma unroll
  for (int i = 0; i < 8; ++i) { r[i] = lo[i]; r[i + 8] = hi[i]; }
  return r;
}

__device__ __forceinline__ bf16x8 cvt8(float4 a, float4 b) {
  bf16x8 v;
  v[0] = (__bf16)a.x; v[1] = (__bf16)a.y; v[2] = (__bf16)a.z; v[3] = (__bf16)a.w;
  v[4] = (__bf16)b.x; v[5] = (__bf16)b.y; v[6] = (__bf16)b.z; v[7] = (__bf16)b.w;
  return v;
}

// ---------------- Router: logits (fp32, exact), softmax, top-2, compaction ---
__global__ __launch_bounds__(256) void moe_router(
    const float* __restrict__ x, const float* __restrict__ gw,
    float* __restrict__ logits_out, int* __restrict__ cnt,
    int* __restrict__ tok_list, float* __restrict__ wt_list,
    int4* __restrict__ tok_rec) {
  const int n = blockIdx.x;
  __shared__ float xs[HID];
  __shared__ float lg[NEXP];
  const int t = threadIdx.x;
  const float* xr = x + (size_t)n * HID;
  ((float4*)xs)[t] = ((const float4*)xr)[t];          // 256 * float4 = 1024
  __syncthreads();
  const int wave = t >> 5, lane = t & 31;
  {
    const float* g = gw + (size_t)wave * HID;          // wave 0..7 -> expert
    float s = 0.f;
#pragma unroll 4
    for (int d = lane; d < HID; d += 32) s += xs[d] * g[d];
    for (int o = 16; o > 0; o >>= 1) s += __shfl_down(s, o, 32);
    if (lane == 0) lg[wave] = s;
  }
  __syncthreads();
  if (t < NEXP) logits_out[(size_t)n * NEXP + t] = lg[t];
  if (t == 0) {
    float m = lg[0];
#pragma unroll
    for (int e = 1; e < NEXP; ++e) m = fmaxf(m, lg[e]);
    float p[NEXP]; float sum = 0.f;
#pragma unroll
    for (int e = 0; e < NEXP; ++e) { p[e] = __expf(lg[e] - m); sum += p[e]; }
    const float inv = 1.f / sum;
#pragma unroll
    for (int e = 0; e < NEXP; ++e) p[e] *= inv;
    int i0 = 0;
#pragma unroll
    for (int e = 1; e < NEXP; ++e) if (p[e] > p[i0]) i0 = e;
    int i1 = (i0 == 0) ? 1 : 0;
#pragma unroll
    for (int e = 0; e < NEXP; ++e) if (e != i0 && p[e] > p[i1]) i1 = e;
    const int s0 = atomicAdd(&cnt[i0], 1);
    const int s1 = atomicAdd(&cnt[i1], 1);
    tok_list[i0 * N_TOK + s0] = n; wt_list[i0 * N_TOK + s0] = p[i0];
    tok_list[i1 * N_TOK + s1] = n; wt_list[i1 * N_TOK + s1] = p[i1];
    tok_rec[n] = make_int4(i0, s0, i1, s1);
  }
}

// ---------------- fp32 -> bf16 conversion of activations ---------------------
__global__ __launch_bounds__(256) void moe_cvt_x(
    const float* __restrict__ x, __bf16* __restrict__ xb, int n8) {
  const int i = blockIdx.x * 256 + threadIdx.x;
  if (i < n8) {
    float4 a = ((const float4*)x)[2 * i];
    float4 b = ((const float4*)x)[2 * i + 1];
    *(bf16x8*)(xb + (size_t)i * 8) = cvt8(a, b);
  }
}

// ---------------- 8-entry prefix sum ----------------------------------------
__global__ void moe_offsets(const int* __restrict__ cnt, int* __restrict__ off) {
  if (threadIdx.x == 0) {
    int acc = 0;
#pragma unroll
    for (int e = 0; e < NEXP; ++e) { off[e] = acc; acc += cnt[e]; }
  }
}

// ---------------- GEMM1: h = silu(x@w1^T) * (x@w3^T), scaled by route wt -----
// block tile: 64 tokens x 128 F. 8 waves on 2x4 grid, wave tile 32x32:
// 2x2 frags each for gate and up (8 v8f accumulators / wave).
__global__ __launch_bounds__(256) void moe_gemm1(
    const __bf16* __restrict__ xb, const float* __restrict__ w1,
    const float* __restrict__ w3, const int* __restrict__ cnt,
    const int* __restrict__ off, const int* __restrict__ tok_list,
    const float* __restrict__ wt_list, __bf16* __restrict__ h) {
  const int e = blockIdx.z;
  const int ne = cnt[e];
  const int mbase = blockIdx.y * 64;
  if (mbase >= ne) return;                  // uniform per block
  const int fbase = blockIdx.x * 128;

  __shared__ __bf16 As[64][32];
  __shared__ __bf16 B1s[128][32];
  __shared__ __bf16 B3s[128][32];
  __shared__ int    row_tok[64];
  __shared__ float  row_wt[64];

  const int t = threadIdx.x;
  if (t < 64) {
    int slot = mbase + t;
    int cs = slot < ne ? slot : ne - 1;     // clamp (rows masked on store)
    row_tok[t] = tok_list[e * N_TOK + cs];
    row_wt[t]  = wt_list[e * N_TOK + cs];
  }
  __syncthreads();                          // row_tok ready

  const int wave = t >> 5, lane = t & 31;
  const int mq = (wave & 1) * 32;           // 0,32
  const int nq = (wave >> 1) * 32;          // 0,32,64,96
  v8f cg[2][2] = {};
  v8f cu[2][2] = {};

  const float* w1e = w1 + (size_t)e * INTER * HID;
  const float* w3e = w3 + (size_t)e * INTER * HID;

  // global-load mapping
  const int alr = t >> 2, alc = (t & 3) * 8;   // A: 64x32 bf16, 8 elems/thread
  const int blr = t >> 1, blc = (t & 1) * 16;  // B: 128x32 f32, 16 elems/thread
  const __bf16* aptr = xb  + (size_t)row_tok[alr] * HID + alc;
  const float*  b1p  = w1e + (size_t)(fbase + blr) * HID + blc;
  const float*  b3p  = w3e + (size_t)(fbase + blr) * HID + blc;

  // frag mapping (contiguous K order, same for A and B)
  const int fr  = lane & 15;
  const int akb = (lane >> 4) * 16;         // A frag: 16 contiguous K
  const int bkb = (lane >> 4) * 16;         // B frag: 16 contiguous K

  // register prefetch pipeline
  bf16x8 ar; float4 b1r[4], b3r[4];
  ar = *(const bf16x8*)aptr;
#pragma unroll
  for (int q = 0; q < 4; ++q) { b1r[q] = *(const float4*)(b1p + 4 * q);
                                b3r[q] = *(const float4*)(b3p + 4 * q); }

  for (int k0 = 0; k0 < HID; k0 += 32) {
    __syncthreads();
    *(bf16x8*)&As[alr][alc]      = ar;
    *(bf16x8*)&B1s[blr][blc]     = cvt8(b1r[0], b1r[1]);
    *(bf16x8*)&B1s[blr][blc + 8] = cvt8(b1r[2], b1r[3]);
    *(bf16x8*)&B3s[blr][blc]     = cvt8(b3r[0], b3r[1]);
    *(bf16x8*)&B3s[blr][blc + 8] = cvt8(b3r[2], b3r[3]);
    __syncthreads();

    if (k0 + 32 < HID) {                    // prefetch next K tile
      const int kn = k0 + 32;
      ar = *(const bf16x8*)(aptr + kn);
#pragma unroll
      for (int q = 0; q < 4; ++q) { b1r[q] = *(const float4*)(b1p + kn + 4 * q);
                                    b3r[q] = *(const float4*)(b3p + kn + 4 * q); }
    }

    v16bf a[2], b1f[2], b3f[2];
#pragma unroll
    for (int mi = 0; mi < 2; ++mi) {
      const int arow = mq + mi * 16 + fr;
      a[mi] = cat8(*(const bf16x8*)&As[arow][akb],
                   *(const bf16x8*)&As[arow][akb + 8]);
    }
#pragma unroll
    for (int ni = 0; ni < 2; ++ni) {
      b1f[ni] = *(const v16bf*)&B1s[nq + ni * 16 + fr][bkb];
      b3f[ni] = *(const v16bf*)&B3s[nq + ni * 16 + fr][bkb];
    }
#pragma unroll
    for (int mi = 0; mi < 2; ++mi)
#pragma unroll
      for (int ni = 0; ni < 2; ++ni) {
        cg[mi][ni] = __builtin_amdgcn_wmma_f32_16x16x32_bf16(
            false, a[mi], false, b1f[ni], (short)0, cg[mi][ni], false, false);
        cu[mi][ni] = __builtin_amdgcn_wmma_f32_16x16x32_bf16(
            false, a[mi], false, b3f[ni], (short)0, cu[mi][ni], false, false);
      }
  }

  // epilogue: h = silu(g)*u * route_wt, store bf16 (compact rows)
  const int colL  = lane & 15;
  const int rhalf = (lane >> 4) * 8;
  const int rowlim = ne - mbase;
  __bf16* hp = h + (size_t)(off[e] + mbase + mq + rhalf) * INTER
                 + (fbase + nq + colL);
#pragma unroll
  for (int mi = 0; mi < 2; ++mi)
#pragma unroll
    for (int v = 0; v < 8; ++v) {
      const int rl = mq + rhalf + mi * 16 + v;     // local row 0..63
      if (rl < rowlim) {
        const float wt = row_wt[rl];
#pragma unroll
        for (int ni = 0; ni < 2; ++ni) {
          const float g = cg[mi][ni][v], u = cu[mi][ni][v];
          const float sig = __builtin_amdgcn_rcpf(1.f + __expf(-g));
          hp[(size_t)(mi * 16 + v) * INTER + ni * 16] = (__bf16)(g * sig * u * wt);
        }
      }
    }
}

// ---------------- GEMM2: part = h @ w2e^T (w2e[f,d] = w2[d, e*F+f]) ----------
// block tile: 64 rows x 128 D. 8 waves on 2x4 grid, wave tile 32x32.
__global__ __launch_bounds__(256) void moe_gemm2(
    const __bf16* __restrict__ h, const float* __restrict__ w2,
    const int* __restrict__ cnt, const int* __restrict__ off,
    float* __restrict__ part) {
  const int e = blockIdx.z;
  const int ne = cnt[e];
  const int mbase = blockIdx.y * 64;
  if (mbase >= ne) return;
  const int dbase = blockIdx.x * 128;
  const int base = off[e];

  __shared__ __bf16 As[64][32];
  __shared__ __bf16 Bs[128][32];

  const int t = threadIdx.x, wave = t >> 5, lane = t & 31;
  const int mq = (wave & 1) * 32;
  const int nq = (wave >> 1) * 32;
  v8f c[2][2] = {};

  const int alr = t >> 2, alc = (t & 3) * 8;
  const int blr = t >> 1, blc = (t & 1) * 16;
  int arow = mbase + alr; if (arow >= ne) arow = ne - 1;
  const __bf16* aptr = h  + (size_t)(base + arow) * INTER + alc;
  const float*  bptr = w2 + (size_t)(dbase + blr) * EF + (size_t)e * INTER + blc;

  const int fr  = lane & 15;
  const int akb = (lane >> 4) * 16;
  const int bkb = (lane >> 4) * 16;

  bf16x8 ar; float4 br[4];
  ar = *(const bf16x8*)aptr;
#pragma unroll
  for (int q = 0; q < 4; ++q) br[q] = *(const float4*)(bptr + 4 * q);

  for (int k0 = 0; k0 < INTER; k0 += 32) {
    __syncthreads();
    *(bf16x8*)&As[alr][alc]     = ar;
    *(bf16x8*)&Bs[blr][blc]     = cvt8(br[0], br[1]);
    *(bf16x8*)&Bs[blr][blc + 8] = cvt8(br[2], br[3]);
    __syncthreads();

    if (k0 + 32 < INTER) {
      const int kn = k0 + 32;
      ar = *(const bf16x8*)(aptr + kn);
#pragma unroll
      for (int q = 0; q < 4; ++q) br[q] = *(const float4*)(bptr + kn + 4 * q);
    }

    v16bf a[2], bf[2];
#pragma unroll
    for (int mi = 0; mi < 2; ++mi) {
      const int arw = mq + mi * 16 + fr;
      a[mi] = cat8(*(const bf16x8*)&As[arw][akb],
                   *(const bf16x8*)&As[arw][akb + 8]);
    }
#pragma unroll
    for (int ni = 0; ni < 2; ++ni)
      bf[ni] = *(const v16bf*)&Bs[nq + ni * 16 + fr][bkb];
#pragma unroll
    for (int mi = 0; mi < 2; ++mi)
#pragma unroll
      for (int ni = 0; ni < 2; ++ni)
        c[mi][ni] = __builtin_amdgcn_wmma_f32_16x16x32_bf16(
            false, a[mi], false, bf[ni], (short)0, c[mi][ni], false, false);
  }

  const int colL  = lane & 15;
  const int rhalf = (lane >> 4) * 8;
  const int rowlim = ne - mbase;
  float* pp = part + (size_t)(base + mbase + mq + rhalf) * HID
                   + (dbase + nq + colL);
#pragma unroll
  for (int mi = 0; mi < 2; ++mi)
#pragma unroll
    for (int v = 0; v < 8; ++v) {
      const int rl = mq + rhalf + mi * 16 + v;
      if (rl < rowlim) {
#pragma unroll
        for (int ni = 0; ni < 2; ++ni)
          pp[(size_t)(mi * 16 + v) * HID + ni * 16] = c[mi][ni][v];
      }
    }
}

// ---------------- Combine: out[n] = part[row(e0,s0)] + part[row(e1,s1)] ------
__global__ __launch_bounds__(256) void moe_combine(
    const float* __restrict__ part, const int* __restrict__ off,
    const int4* __restrict__ rec, float* __restrict__ out) {
  const int n = blockIdx.x;
  const int4 r = rec[n];
  const float* p0 = part + (size_t)(off[r.x] + r.y) * HID;
  const float* p1 = part + (size_t)(off[r.z] + r.w) * HID;
  float* o = out + (size_t)n * HID;
  const int t = threadIdx.x;                  // 256 threads * float4 = 1024
  float4 a = ((const float4*)p0)[t];
  float4 b = ((const float4*)p1)[t];
  float4 s; s.x = a.x + b.x; s.y = a.y + b.y; s.z = a.z + b.z; s.w = a.w + b.w;
  ((float4*)o)[t] = s;
}

// ---------------- launch -----------------------------------------------------
extern "C" void kernel_launch(void* const* d_in, const int* in_sizes, int n_in,
                              void* d_out, int out_size, void* d_ws, size_t ws_size,
                              hipStream_t stream) {
  const float* x  = (const float*)d_in[0];   // [2,2048,1024]
  const float* gw = (const float*)d_in[1];   // [8,1024]
  const float* w1 = (const float*)d_in[2];   // [8*2816,1024]
  const float* w3 = (const float*)d_in[3];   // [8*2816,1024]
  const float* w2 = (const float*)d_in[4];   // [1024, 8*2816]
  float* out = (float*)d_out;                // final [4096*1024] ++ logits [4096*8]

  char* ws = (char*)d_ws;
  int*    cnt  = (int*)(ws + 0);                       // 8 ints
  int*    off  = (int*)(ws + 128);                     // 8 ints
  int4*   rec  = (int4*)(ws + 256);                    // 4096 * int4 = 64KB
  int*    tok  = (int*)(ws + 65792);                   // 8*4096 ints = 128KB
  float*  wt   = (float*)(ws + 196864);                // 8*4096 f32 = 128KB
  __bf16* xb   = (__bf16*)(ws + 393216);               // 4096*1024 bf16 = 8MB
  __bf16* hbuf = (__bf16*)(ws + 8781824);              // 8192*2816 bf16 = 46MB
  float*  part = (float*)(ws + 54919168);              // 8192*1024 f32 = 33.5MB

  float* logits = out + (size_t)N_TOK * HID;

  hipMemsetAsync(cnt, 0, NEXP * sizeof(int), stream);
  moe_router<<<N_TOK, 256, 0, stream>>>(x, gw, logits, cnt, tok, wt, rec);
  moe_cvt_x<<<(N_TOK * HID / 8 + 255) / 256, 256, 0, stream>>>(x, xb, N_TOK * HID / 8);
  moe_offsets<<<1, 32, 0, stream>>>(cnt, off);
  moe_gemm1<<<dim3(INTER / 128, N_TOK / 64, NEXP), 256, 0, stream>>>(
      xb, w1, w3, cnt, off, tok, wt, hbuf);
  moe_gemm2<<<dim3(HID / 128, N_TOK / 64, NEXP), 256, 0, stream>>>(
      hbuf, w2, cnt, off, part);
  moe_combine<<<N_TOK, 256, 0, stream>>>(part, off, rec, out);
}
